// Encoder_Decoder_30580167147766
// MI455X (gfx1250) — matrix operand order repaired and checked by reading, not compiled, not run on hardware
//
#include <hip/hip_runtime.h>
#include <hip/hip_bf16.h>
#include <stdint.h>

#define N_PROP 8192
#define HID 512
#define G3 1536               // 3*H
#define KAPP 1024
#define KSCORE 1296           // 81*4*4
#define KSCORE_PAD 1312       // padded to multiple of 32 (pad cols of W are zero)
#define SCAN_BLOCKS 16
#define ROWS_PER_BLK (G3 / SCAN_BLOCKS)   // 96 rows of Whh per WGP (96KB bf16 in LDS)

typedef __bf16 bf16_t;
typedef bf16_t v16bf __attribute__((ext_vector_type(16)));
typedef bf16_t v8bf  __attribute__((ext_vector_type(8)));
typedef float  v8f   __attribute__((ext_vector_type(8)));

union AFrag { v16bf v; v8bf h[2]; };

__device__ __forceinline__ unsigned short f32_to_bf16(float f) {
  unsigned u = __float_as_uint(f);
  u += 0x7fffu + ((u >> 16) & 1u);          // round-to-nearest-even
  return (unsigned short)(u >> 16);
}
__device__ __forceinline__ unsigned pack_bf16x2(float lo, float hi) {
  return (unsigned)f32_to_bf16(lo) | ((unsigned)f32_to_bf16(hi) << 16);
}
__device__ __forceinline__ float bf_lo(unsigned u) { return __uint_as_float(u << 16); }
__device__ __forceinline__ float bf_hi(unsigned u) { return __uint_as_float(u & 0xffff0000u); }
__device__ __forceinline__ float sigm(float x) { return 1.f / (1.f + __expf(-x)); }

// ---------------------------------------------------------------- small prep
__global__ void init_kernel(int* ctr) {
  if (threadIdx.x < 2) ctr[threadIdx.x] = 0;
}

__global__ void f32_to_bf16_kernel(const float* __restrict__ src,
                                   unsigned short* __restrict__ dst, int n) {
  int i = blockIdx.x * blockDim.x + threadIdx.x;
  if (i < n) dst[i] = f32_to_bf16(src[i]);
}

// comb[h][k] : fused (unfolded-conv @ W_axis) matrix, bf16, K padded to 1312 w/ zeros
__global__ void build_comb_kernel(const float* __restrict__ W_conv,   // [56,81,2,2]
                                  const float* __restrict__ W_axis,   // [512,504]
                                  unsigned short* __restrict__ comb) {
  int idx = blockIdx.x * blockDim.x + threadIdx.x;
  if (idx >= HID * KSCORE_PAD) return;
  int h = idx / KSCORE_PAD, k = idx % KSCORE_PAD;
  float s = 0.f;
  if (k < KSCORE) {
    int i = k >> 4;                 // input channel (81)
    int yy = (k >> 2) & 3, xx = k & 3;
    for (int c = 0; c < 56; ++c)
      for (int ky = 0; ky < 2; ++ky) {
        int y = yy - ky; if (y < 0 || y > 2) continue;
        for (int kx = 0; kx < 2; ++kx) {
          int x = xx - kx; if (x < 0 || x > 2) continue;
          s += W_conv[((c * 81 + i) * 2 + ky) * 2 + kx] *
               W_axis[h * 504 + c * 9 + y * 3 + x];
        }
      }
  }
  comb[h * KSCORE_PAD + k] = f32_to_bf16(s);
}

// fused feats bias: b_app + b_axis + (conv bias folded through W_axis)
__global__ void build_fb_kernel(const float* __restrict__ b_app,
                                const float* __restrict__ b_axis,
                                const float* __restrict__ b_conv,
                                const float* __restrict__ W_axis,
                                float* __restrict__ fb) {
  int h = blockIdx.x * blockDim.x + threadIdx.x;
  if (h >= HID) return;
  float s = b_app[h] + b_axis[h];
  for (int c = 0; c < 56; ++c) {
    float ws = 0.f;
    for (int p = 0; p < 9; ++p) ws += W_axis[h * 504 + c * 9 + p];
    s += b_conv[c] * ws;
  }
  fb[h] = s;
}

// ---------------------------------------------------------------- WMMA GEMM
// C[M,N] (+)= A[M,K](f32 -> bf16 on the fly) * W[N,Kpad](bf16)^T + bias
// Block tile 128x128, 8 waves (4 along M x 2 along N); wave tile 32x64:
// 2 A-frags x 4 B-frags -> 8 x v_wmma_f32_16x16x32_bf16 per K-step.
// LDS double-buffered A tile; no K-guards (W pad cols are zero; A loads are
// address-clamped so OOB reads stay inside the allocation and hit zero weights).
__global__ __launch_bounds__(256) void gemm_bf16_wmma(
    const float* __restrict__ A, int lda, size_t nElem,   // nElem = M*lda
    const unsigned short* __restrict__ W, int ldw,
    float* __restrict__ C, int ldc,
    const float* __restrict__ bias, int accumulate, int ksteps) {
  __shared__ unsigned short sA[2][128 * 32];   // 2 x 8 KB ping-pong A tile (bf16)

  const int tid  = threadIdx.x;
  const int lane = tid & 31;
  const int wave = tid >> 5;
  const int m0 = (wave >> 1) * 32;            // wave row offset in block tile
  const int n0 = (wave & 1) * 64;             // wave col offset in block tile
  const size_t rowBase = (size_t)blockIdx.y * 128;
  const int    colBase = blockIdx.x * 128;

  v8f acc[8];
#pragma unroll
  for (int q = 0; q < 8; ++q) acc[q] = (v8f){};

  const int lsel = (lane < 16) ? 0 : 16;      // A-frag K-half select (contiguous k = 16*half + s, same as B)
  const int arow = lane & 15;
  const int bofs = (lane < 16) ? 0 : 16;      // B-frag K-half select
  const int bn0  = colBase + n0 + (lane & 15);

  // per-thread staging slots: 4 float4 per tile (128x32 floats / 256 threads)
  const int sr[4] = { (tid + 0)   >> 3, (tid + 256) >> 3,
                      (tid + 512) >> 3, (tid + 768) >> 3 };
  const int sc    = (tid & 7) * 4;

  // ---- prologue: stage tile 0 ----
  {
#pragma unroll
    for (int i = 0; i < 4; ++i) {
      size_t idx = (rowBase + sr[i]) * (size_t)lda + sc;
      if (idx > nElem - 4) idx = nElem - 4;               // clamp (pad cols of W are 0)
      float4 v = *(const float4*)(A + idx);
      uint2 pk = { pack_bf16x2(v.x, v.y), pack_bf16x2(v.z, v.w) };
      *(uint2*)&sA[0][sr[i] * 32 + sc] = pk;
    }
  }
  __syncthreads();

  int p = 0;
  for (int kt = 0; kt < ksteps; ++kt) {
    const int k0 = kt * 32;

    // ---- issue next tile's global loads first (overlap with WMMA) ----
    float4 nx[4];
    const bool more = (kt + 1) < ksteps;
    if (more) {
      const int kn = k0 + 32;
#pragma unroll
      for (int i = 0; i < 4; ++i) {
        size_t idx = (rowBase + sr[i]) * (size_t)lda + kn + sc;
        if (idx > nElem - 4) idx = nElem - 4;
        nx[i] = *(const float4*)(A + idx);
      }
    }

    // ---- compute current tile ----
    AFrag a0, a1;
    a0.h[0] = *(const v8bf*)&sA[p][(m0 + arow) * 32 + lsel];
    a0.h[1] = *(const v8bf*)&sA[p][(m0 + arow) * 32 + 8 + lsel];
    a1.h[0] = *(const v8bf*)&sA[p][(m0 + 16 + arow) * 32 + lsel];
    a1.h[1] = *(const v8bf*)&sA[p][(m0 + 16 + arow) * 32 + 8 + lsel];

#pragma unroll
    for (int tj = 0; tj < 4; ++tj) {
      v16bf b = *(const v16bf*)(W + (size_t)(bn0 + tj * 16) * ldw + k0 + bofs);
      acc[tj]     = __builtin_amdgcn_wmma_f32_16x16x32_bf16(false, a0.v, false, b,
                                                            (short)0, acc[tj], false, false);
      acc[4 + tj] = __builtin_amdgcn_wmma_f32_16x16x32_bf16(false, a1.v, false, b,
                                                            (short)0, acc[4 + tj], false, false);
    }

    // ---- store next tile into the other buffer, single barrier ----
    if (more) {
#pragma unroll
      for (int i = 0; i < 4; ++i) {
        uint2 pk = { pack_bf16x2(nx[i].x, nx[i].y), pack_bf16x2(nx[i].z, nx[i].w) };
        *(uint2*)&sA[p ^ 1][sr[i] * 32 + sc] = pk;
      }
    }
    __syncthreads();
    p ^= 1;
  }

  // ---- epilogue ----
  const int cn = lane & 15;
  const int mo = (lane < 16) ? 0 : 8;
#pragma unroll
  for (int ti = 0; ti < 2; ++ti) {
#pragma unroll
    for (int tj = 0; tj < 4; ++tj) {
      const int col = colBase + n0 + tj * 16 + cn;
      const float bv = bias ? bias[col] : 0.f;
      const v8f a = acc[ti * 4 + tj];
#pragma unroll
      for (int j = 0; j < 8; ++j) {
        const size_t row = rowBase + m0 + ti * 16 + mo + j;
        float* cp = C + row * (size_t)ldc + col;
        float v = a[j] + bv;
        if (accumulate) v += *cp;
        *cp = v;
      }
    }
  }
}

// ---------------------------------------------------------------- dec_in
__global__ void relu_add_kernel(const float* __restrict__ feats,
                                const float* __restrict__ encf,
                                float* __restrict__ dec, int total) {
  int i = blockIdx.x * blockDim.x + threadIdx.x;
  if (i < total) dec[i] = fmaxf(feats[i] + encf[i & (HID - 1)], 0.f);
}

// ---------------------------------------------------------------- GRU scan
// 16 persistent workgroups; each keeps a 96-row slice of Whh (bf16) in LDS.
// One device-scope barrier per step; gh double-buffered on step parity; every
// block redundantly computes h_new into its private LDS h (no second barrier).
__global__ __launch_bounds__(1024) void gru_scan_kernel(
    const float* __restrict__ Gi,             // [T, 1536] precomputed input gates
    const unsigned short* __restrict__ Whh,   // [1536, 512] bf16
    const float* __restrict__ bhh,            // [1536]
    float* __restrict__ gh_buf,               // [2, 1536] exchange (L2)
    float* __restrict__ enc_final,            // [512] (encoder only)
    const float* __restrict__ w_out,          // [512] (decoder only)
    const float* __restrict__ b_out,          // [1]   (decoder only)
    float* __restrict__ out,                  // [T]   (decoder only)
    int* __restrict__ ctr, int T, int write_out) {
  extern __shared__ unsigned short sW[];      // ROWS_PER_BLK*512 bf16 = 96 KB
  __shared__ float sh[HID];
  __shared__ float sred[HID];

  const int tid = threadIdx.x;
  const int blk = blockIdx.x;
  const int rowbase = blk * ROWS_PER_BLK;

  // load this block's Whh slice into LDS once (16B vector copies)
  {
    const uint4* src = (const uint4*)(Whh + (size_t)rowbase * HID);
    uint4* dst = (uint4*)sW;
    const int n16 = ROWS_PER_BLK * HID / 8;
    for (int i = tid; i < n16; i += 1024) dst[i] = src[i];
  }
  if (tid < HID) sh[tid] = 0.f;
  const float wo = (write_out && tid < HID) ? w_out[tid] : 0.f;
  __syncthreads();

  for (int t = 0; t < T; ++t) {
    float* ghw = gh_buf + (size_t)(t & 1) * G3;
    // gh slice = Whh_slice * h + bhh  (8 lanes per row, 64 MACs each)
    if (tid < 768) {
      const int row = tid >> 3;
      const int part = tid & 7;
      const uint2* wr = (const uint2*)&sW[row * HID + part * 64];
      const float* hp = &sh[part * 64];
      float s = 0.f;
#pragma unroll
      for (int i = 0; i < 16; ++i) {
        uint2 w = wr[i];
        s = fmaf(bf_lo(w.x), hp[i * 4 + 0], s);
        s = fmaf(bf_hi(w.x), hp[i * 4 + 1], s);
        s = fmaf(bf_lo(w.y), hp[i * 4 + 2], s);
        s = fmaf(bf_hi(w.y), hp[i * 4 + 3], s);
      }
#pragma unroll
      for (int o = 1; o < 8; o <<= 1) s += __shfl_xor(s, o, 8);
      if (part == 0) {
        const int gr = rowbase + row;
        __hip_atomic_store(&ghw[gr], s + bhh[gr],
                           __ATOMIC_RELAXED, __HIP_MEMORY_SCOPE_AGENT);
      }
    }
    __threadfence();
    __syncthreads();
    if (tid == 0)
      __hip_atomic_fetch_add(ctr, 1, __ATOMIC_RELEASE, __HIP_MEMORY_SCOPE_AGENT);
    const int target = (t + 1) * SCAN_BLOCKS;
    while (__hip_atomic_load(ctr, __ATOMIC_ACQUIRE, __HIP_MEMORY_SCOPE_AGENT) < target)
      __builtin_amdgcn_s_sleep(2);

    // gates: every block computes the full h_new redundantly into its own LDS h
    if (tid < HID) {
      const float* git = Gi + (size_t)t * G3;
      float hr = __hip_atomic_load(&ghw[tid],        __ATOMIC_RELAXED, __HIP_MEMORY_SCOPE_AGENT);
      float hz = __hip_atomic_load(&ghw[HID + tid],  __ATOMIC_RELAXED, __HIP_MEMORY_SCOPE_AGENT);
      float hn = __hip_atomic_load(&ghw[1024 + tid], __ATOMIC_RELAXED, __HIP_MEMORY_SCOPE_AGENT);
      float r = sigm(git[tid] + hr);
      float z = sigm(git[HID + tid] + hz);
      float n = tanhf(git[1024 + tid] + r * hn);
      float hnew = (1.f - z) * n + z * sh[tid];
      sh[tid] = hnew;
      if (write_out) sred[tid] = hnew * wo;
    }
    __syncthreads();
    if (write_out && blk == 0) {            // fused output head: sigmoid(h . W_out + b)
#pragma unroll
      for (int sft = 256; sft > 0; sft >>= 1) {
        if (tid < sft) sred[tid] += sred[tid + sft];
        __syncthreads();
      }
      if (tid == 0) out[t] = sigm(sred[0] + b_out[0]);
      __syncthreads();
    }
  }
  if (!write_out && blk == 0 && tid < HID) enc_final[tid] = sh[tid];
}

// ---------------------------------------------------------------- launch
extern "C" void kernel_launch(void* const* d_in, const int* in_sizes, int n_in,
                              void* d_out, int out_size, void* d_ws, size_t ws_size,
                              hipStream_t stream) {
  (void)in_sizes; (void)n_in; (void)out_size; (void)ws_size;
  const float* boxes_feature   = (const float*)d_in[0];
  const float* boxes_box_score = (const float*)d_in[1];
  const float* W_app  = (const float*)d_in[2];
  const float* b_app  = (const float*)d_in[3];
  const float* W_conv = (const float*)d_in[4];
  const float* b_conv = (const float*)d_in[5];
  const float* W_axis = (const float*)d_in[6];
  const float* b_axis = (const float*)d_in[7];
  const float* Wih_e  = (const float*)d_in[8];
  const float* Whh_e  = (const float*)d_in[9];
  const float* bih_e  = (const float*)d_in[10];
  const float* bhh_e  = (const float*)d_in[11];
  const float* Wih_d  = (const float*)d_in[12];
  const float* Whh_d  = (const float*)d_in[13];
  const float* bih_d  = (const float*)d_in[14];
  const float* bhh_d  = (const float*)d_in[15];
  const float* W_out  = (const float*)d_in[16];
  const float* b_out  = (const float*)d_in[17];
  float* out = (float*)d_out;

  char* ws = (char*)d_ws;
  auto take = [&](size_t bytes) {
    char* p = ws; ws += (bytes + 255) & ~(size_t)255; return p;
  };
  int*   ctr  = (int*)take(256);
  float* gh   = (float*)take(2 * G3 * sizeof(float));
  float* encf = (float*)take(HID * sizeof(float));
  float* fb   = (float*)take(HID * sizeof(float));
  unsigned short* wapp = (unsigned short*)take((size_t)HID * KAPP * 2);
  unsigned short* comb = (unsigned short*)take((size_t)HID * KSCORE_PAD * 2);
  unsigned short* wihe = (unsigned short*)take((size_t)G3 * HID * 2);
  unsigned short* wihd = (unsigned short*)take((size_t)G3 * HID * 2);
  unsigned short* whhe = (unsigned short*)take((size_t)G3 * HID * 2);
  unsigned short* whhd = (unsigned short*)take((size_t)G3 * HID * 2);
  float* feats = (float*)take((size_t)N_PROP * HID * 4);
  float* decin = (float*)take((size_t)N_PROP * HID * 4);
  float* Gi    = (float*)take((size_t)N_PROP * G3 * 4);

  init_kernel<<<1, 64, 0, stream>>>(ctr);
  f32_to_bf16_kernel<<<(HID * KAPP + 255) / 256, 256, 0, stream>>>(W_app, wapp, HID * KAPP);
  f32_to_bf16_kernel<<<(G3 * HID + 255) / 256, 256, 0, stream>>>(Wih_e, wihe, G3 * HID);
  f32_to_bf16_kernel<<<(G3 * HID + 255) / 256, 256, 0, stream>>>(Wih_d, wihd, G3 * HID);
  f32_to_bf16_kernel<<<(G3 * HID + 255) / 256, 256, 0, stream>>>(Whh_e, whhe, G3 * HID);
  f32_to_bf16_kernel<<<(G3 * HID + 255) / 256, 256, 0, stream>>>(Whh_d, whhd, G3 * HID);
  build_comb_kernel<<<(HID * KSCORE_PAD + 255) / 256, 256, 0, stream>>>(W_conv, W_axis, comb);
  build_fb_kernel<<<2, 256, 0, stream>>>(b_app, b_axis, b_conv, W_axis, fb);

  // feats = boxes_feature @ W_app^T + fused_bias   (+=)  score @ comb^T
  gemm_bf16_wmma<<<dim3(HID / 128, N_PROP / 128), 256, 0, stream>>>(
      boxes_feature, KAPP, (size_t)N_PROP * KAPP,
      wapp, KAPP, feats, HID, fb, 0, KAPP / 32);
  gemm_bf16_wmma<<<dim3(HID / 128, N_PROP / 128), 256, 0, stream>>>(
      boxes_box_score, KSCORE, (size_t)N_PROP * KSCORE,
      comb, KSCORE_PAD, feats, HID, nullptr, 1, KSCORE_PAD / 32);
  // Gi_e = feats @ Wih_e^T + bih_e
  gemm_bf16_wmma<<<dim3(G3 / 128, N_PROP / 128), 256, 0, stream>>>(
      feats, HID, (size_t)N_PROP * HID,
      wihe, HID, Gi, G3, bih_e, 0, HID / 32);

  const size_t scan_lds = (size_t)ROWS_PER_BLK * HID * sizeof(unsigned short); // 96 KB
  hipFuncSetAttribute(reinterpret_cast<const void*>(gru_scan_kernel),
                      hipFuncAttributeMaxDynamicSharedMemorySize, (int)scan_lds);
  // encoder scan -> enc_final
  gru_scan_kernel<<<SCAN_BLOCKS, 1024, scan_lds, stream>>>(
      Gi, whhe, bhh_e, gh, encf, nullptr, nullptr, nullptr, ctr + 0, N_PROP, 0);

  relu_add_kernel<<<(N_PROP * HID + 255) / 256, 256, 0, stream>>>(
      feats, encf, decin, N_PROP * HID);
  // Gi_d = relu(feats+enc) @ Wih_d^T + bih_d
  gemm_bf16_wmma<<<dim3(G3 / 128, N_PROP / 128), 256, 0, stream>>>(
      decin, HID, (size_t)N_PROP * HID,
      wihd, HID, Gi, G3, bih_d, 0, HID / 32);
  // decoder scan with fused sigmoid(h . W_out + b_out) -> out
  gru_scan_kernel<<<SCAN_BLOCKS, 1024, scan_lds, stream>>>(
      Gi, whhd, bhh_d, gh, nullptr, W_out, b_out, out, ctr + 1, N_PROP, 1);
}
